// MotionPrediction_61074434949740
// MI455X (gfx1250) — matrix operand order repaired and checked by reading, not compiled, not run on hardware
//
#include <hip/hip_runtime.h>
#include <hip/hip_bf16.h>

typedef __attribute__((ext_vector_type(16))) _Float16 v16h;
typedef __attribute__((ext_vector_type(8)))  _Float16 v8h;
typedef __attribute__((ext_vector_type(8)))  float    v8f;

#define B_DIM 8192
#define E_DIM 8
#define I_DIM 512
#define H_DIM 1024
#define O_DIM 512

// ---------------------------------------------------------------------------
// f32 -> f16 conversion (vectorized, 8 elements / thread)
// ---------------------------------------------------------------------------
__global__ void cvt_f32_to_f16(const float* __restrict__ s, _Float16* __restrict__ d, int n) {
  int i = (blockIdx.x * blockDim.x + threadIdx.x) * 8;
  if (i + 8 <= n) {
    const float4 v0 = *(const float4*)(s + i);
    const float4 v1 = *(const float4*)(s + i + 4);
    v8h h;
    h[0] = (_Float16)v0.x; h[1] = (_Float16)v0.y;
    h[2] = (_Float16)v0.z; h[3] = (_Float16)v0.w;
    h[4] = (_Float16)v1.x; h[5] = (_Float16)v1.y;
    h[6] = (_Float16)v1.z; h[7] = (_Float16)v1.w;
    *(v8h*)(d + i) = h;
  } else {
    for (; i < n; ++i) d[i] = (_Float16)s[i];
  }
}

// ---------------------------------------------------------------------------
// WMMA fragment loaders (contiguous-K order shared by A and B, wave32).
// A 16x32 f16: lane l holds row M=l&15; one contiguous 32B run at K = 16*(l>>4).
// B 32x16 f16: lane l holds col N=l&15; one contiguous 32B run at K = 16*(l>>4).
// Pointers are pre-offset per lane; only k0 varies in the inner loop so the
// compiler can fold K-steps into instruction immediate offsets.
// ---------------------------------------------------------------------------
__device__ __forceinline__ v16h load_a(const _Float16* __restrict__ p) {
  v8h lo = *(const v8h*)p;
  v8h hi = *(const v8h*)(p + 8);
  return __builtin_shufflevector(lo, hi, 0, 1, 2, 3, 4, 5, 6, 7,
                                         8, 9, 10, 11, 12, 13, 14, 15);
}

__device__ __forceinline__ v16h load_b(const _Float16* __restrict__ p) {
  return *(const v16h*)p;
}

#define WMMA_F16(a, b, c) \
  __builtin_amdgcn_wmma_f32_16x16x32_f16(false, (a), false, (b), (short)0, (c), false, false)

// ---------------------------------------------------------------------------
// Fused expert-GEMM layer:
//   out[b,n] = act( sum_e gate[b,e] * ( sum_k A[b,k]*Wexp[e,n,k] + bias[e,n] ) )
// Block: 256 threads = 8 waves; 128x64 output tile; wave = 32x32 (2x2 WMMA).
// Experts processed in PAIRS so each A fragment feeds 4 WMMAs (12 B128 loads
// per 8 WMMAs); gate contraction fused in-register.
// ---------------------------------------------------------------------------
template <bool APPLY_ELU, bool OUT_HALF>
__global__ __launch_bounds__(256)
void moe_stage(const _Float16* __restrict__ A,      // [M,K]   f16
               const _Float16* __restrict__ Wexp,   // [E,N,K] f16
               const float* __restrict__ bias,      // [E,N]   f32
               const float* __restrict__ gate,      // [M,E]   f32
               void* __restrict__ outp,             // [M,N]   f16 or f32
               int M, int N, int K) {
  __shared__ __align__(16) float gl[128 * E_DIM];

  const int tid  = threadIdx.x;
  const int lane = tid & 31;
  const int lm   = lane & 15;
  const int lh   = lane >> 4;
  const int wave = tid >> 5;
  const int wm   = wave & 3;   // 4 waves along M
  const int wn   = wave >> 2;  // 2 waves along N
  const int mblk = blockIdx.y * 128;
  const int nblk = blockIdx.x * 64;

  // Stage the 128x8 gate block (contiguous in memory) into LDS: 1 float4/thread.
  ((float4*)gl)[tid] = ((const float4*)(gate + (size_t)mblk * E_DIM))[tid];
  __syncthreads();

  const int row0 = mblk + wm * 32;
  const int col0 = nblk + wn * 32;

  // Per-lane invariant base pointers (A fragments are expert-invariant).
  const _Float16* __restrict__ pa0 = A + (size_t)(row0 + lm) * K + lh * 16;
  const _Float16* __restrict__ pa1 = pa0 + (size_t)16 * K;
  const _Float16* __restrict__ pfa = A + (size_t)(row0 + lane) * K;  // prefetch: 32 rows

  v8f cmb[2][2] = {};  // gate-combined accumulator (f32)

  for (int e = 0; e < E_DIM; e += 2) {
    const _Float16* __restrict__ Be0 = Wexp + (size_t)e * N * K;
    const _Float16* __restrict__ Be1 = Be0 + (size_t)N * K;
    const _Float16* __restrict__ pb00 = Be0 + (size_t)(col0 + lm) * K + lh * 16;
    const _Float16* __restrict__ pb01 = pb00 + (size_t)16 * K;
    const _Float16* __restrict__ pb10 = Be1 + (size_t)(col0 + lm) * K + lh * 16;
    const _Float16* __restrict__ pb11 = pb10 + (size_t)16 * K;
    const _Float16* __restrict__ pfb0 = Be0 + (size_t)(col0 + lane) * K;  // prefetch
    const _Float16* __restrict__ pfb1 = Be1 + (size_t)(col0 + lane) * K;

    v8f acc0[2][2] = {};
    v8f acc1[2][2] = {};

#pragma unroll 4
    for (int k0 = 0; k0 < K; k0 += 32) {
      if ((k0 & 63) == 0) {  // gfx1250 global_prefetch_b8, near-cache locality
        __builtin_prefetch(pfa  + k0 + 256, 0, 3);
        __builtin_prefetch(pfb0 + k0 + 256, 0, 3);
        __builtin_prefetch(pfb1 + k0 + 256, 0, 3);
      }
      v16h a0  = load_a(pa0 + k0);
      v16h a1  = load_a(pa1 + k0);
      v16h b00 = load_b(pb00 + k0);
      v16h b01 = load_b(pb01 + k0);
      acc0[0][0] = WMMA_F16(a0, b00, acc0[0][0]);
      acc0[0][1] = WMMA_F16(a0, b01, acc0[0][1]);
      acc0[1][0] = WMMA_F16(a1, b00, acc0[1][0]);
      acc0[1][1] = WMMA_F16(a1, b01, acc0[1][1]);
      v16h b10 = load_b(pb10 + k0);
      v16h b11 = load_b(pb11 + k0);
      acc1[0][0] = WMMA_F16(a0, b10, acc1[0][0]);
      acc1[0][1] = WMMA_F16(a0, b11, acc1[0][1]);
      acc1[1][0] = WMMA_F16(a1, b10, acc1[1][0]);
      acc1[1][1] = WMMA_F16(a1, b11, acc1[1][1]);
    }

    // Epilogue: cmb += g_e*(acc_e + bias_e) for both experts of the pair.
    // C layout: N = lane&15 (fixed/lane); M = vgpr_index + 8*(lane>>4) per tile.
#pragma unroll
    for (int tn = 0; tn < 2; ++tn) {
      const int nidx = col0 + tn * 16 + lm;
      const float bn0 = bias[(size_t)e * N + nidx];
      const float bn1 = bias[(size_t)(e + 1) * N + nidx];
#pragma unroll
      for (int tm = 0; tm < 2; ++tm) {
#pragma unroll
        for (int r = 0; r < 8; ++r) {
          const int mloc = wm * 32 + tm * 16 + lh * 8 + r;  // row within block
          const float g0 = gl[mloc * E_DIM + e];
          const float g1 = gl[mloc * E_DIM + e + 1];
          cmb[tm][tn][r] += g0 * (acc0[tm][tn][r] + bn0) +
                            g1 * (acc1[tm][tn][r] + bn1);
        }
      }
    }
  }

  // Activation + store.
#pragma unroll
  for (int tn = 0; tn < 2; ++tn) {
    const int gc = col0 + tn * 16 + lm;
#pragma unroll
    for (int tm = 0; tm < 2; ++tm) {
#pragma unroll
      for (int r = 0; r < 8; ++r) {
        const int gr = row0 + tm * 16 + lh * 8 + r;
        float v = cmb[tm][tn][r];
        if (APPLY_ELU) v = (v > 0.0f) ? v : (__expf(v) - 1.0f);
        if (OUT_HALF)
          ((_Float16*)outp)[(size_t)gr * N + gc] = (_Float16)v;
        else
          ((float*)outp)[(size_t)gr * N + gc] = v;
      }
    }
  }
}

// ---------------------------------------------------------------------------
// Launch: 4 conversion passes, then the 3 fused layers.
// Workspace layout (72 MiB total):
//   [0,8)MiB Xh | [8,16) W1h | [16,32) W2h | [32,40) W3h | [40,56) mid1 | [56,72) mid2
// ---------------------------------------------------------------------------
extern "C" void kernel_launch(void* const* d_in, const int* in_sizes, int n_in,
                              void* d_out, int out_size, void* d_ws, size_t ws_size,
                              hipStream_t stream) {
  (void)in_sizes; (void)n_in; (void)out_size; (void)ws_size;

  const float* X  = (const float*)d_in[0];
  const float* Wg = (const float*)d_in[1];
  const float* W1 = (const float*)d_in[2];
  const float* b1 = (const float*)d_in[3];
  const float* W2 = (const float*)d_in[4];
  const float* b2 = (const float*)d_in[5];
  const float* W3 = (const float*)d_in[6];
  const float* b3 = (const float*)d_in[7];

  char* ws = (char*)d_ws;
  const size_t MB = (size_t)1 << 20;
  _Float16* Xh   = (_Float16*)(ws + 0 * MB);
  _Float16* W1h  = (_Float16*)(ws + 8 * MB);
  _Float16* W2h  = (_Float16*)(ws + 16 * MB);
  _Float16* W3h  = (_Float16*)(ws + 32 * MB);
  _Float16* mid1 = (_Float16*)(ws + 40 * MB);
  _Float16* mid2 = (_Float16*)(ws + 56 * MB);

  auto cvt = [&](const float* s, _Float16* d, int n) {
    int blocks = (n / 8 + 255) / 256;
    cvt_f32_to_f16<<<blocks, 256, 0, stream>>>(s, d, n);
  };
  cvt(X,  Xh,  B_DIM * I_DIM);
  cvt(W1, W1h, E_DIM * H_DIM * I_DIM);
  cvt(W2, W2h, E_DIM * H_DIM * H_DIM);
  cvt(W3, W3h, E_DIM * O_DIM * H_DIM);

  dim3 blk(256);
  moe_stage<true,  true ><<<dim3(H_DIM / 64, B_DIM / 128), blk, 0, stream>>>(
      Xh,   W1h, b1, Wg, mid1, B_DIM, H_DIM, I_DIM);
  moe_stage<true,  true ><<<dim3(H_DIM / 64, B_DIM / 128), blk, 0, stream>>>(
      mid1, W2h, b2, Wg, mid2, B_DIM, H_DIM, H_DIM);
  moe_stage<false, false><<<dim3(O_DIM / 64, B_DIM / 128), blk, 0, stream>>>(
      mid2, W3h, b3, Wg, d_out, B_DIM, O_DIM, H_DIM);
}
